// LSTMContinuousCritic_70102456205976
// MI455X (gfx1250) — matrix operand order repaired and checked by reading, not compiled, not run on hardware
//
#include <hip/hip_runtime.h>

typedef __attribute__((ext_vector_type(16))) __bf16 v16bf;
typedef __attribute__((ext_vector_type(8)))  __bf16 v8bf;
typedef __attribute__((ext_vector_type(8)))  float  v8f;
typedef __attribute__((ext_vector_type(4)))  float  v4f;

namespace {
constexpr int B = 1024, S = 512, D = 128, H = 64, G = 256;  // G = 4H
constexpr int WAVES = 4;           // waves per workgroup
constexpr int RPW   = 16;          // batch rows per wave (= WMMA M)
constexpr int RPWG  = WAVES * RPW; // 64 rows per workgroup

// ---- LDS byte layout (dynamic LDS; 320KB/WGP on CDNA5 makes this easy) ----
constexpr int OFF_WIN  = 0;                  // W_in  [H][D] bf16 : 16384 B
constexpr int OFF_WIH  = OFF_WIN + H * D * 2; // W_ih  [G][H] bf16 : 32768 B
constexpr int OFF_WHH  = OFF_WIH + G * H * 2; // W_hh  [G][H] bf16 : 32768 B
constexpr int OFF_BIN  = OFF_WHH + G * H * 2; // b_in  [H] f32
constexpr int OFF_BSUM = OFF_BIN + H * 4;     // b_ih+b_hh [G] f32
constexpr int OFF_WOUT = OFF_BSUM + G * 4;    // W_out [H] f32
constexpr int OFF_SCR  = OFF_WOUT + H * 4;    // per-wave scratch: h(2KB)+z(2KB)
constexpr int SCR_PER_WAVE = 2 * RPW * H * 2; // 4096 B
constexpr int LDS_BYTES = OFF_SCR + WAVES * SCR_PER_WAVE; // 99840 B
} // namespace

static __device__ inline v16bf cat8(v8bf lo, v8bf hi) {
  union { v16bf v; v8bf h[2]; } u;
  u.h[0] = lo; u.h[1] = hi;
  return u.v;
}

// B operand tile (K=32 x N=16, bf16). Weights stored row-major [col][k] in LDS,
// so B[k][n] = W[cb+n][kb+k]; lane elems 0..7 = K = kb+off+0..7, elems 8..15 = +16 (off = 0/8).
static __device__ inline v16bf loadB(const __bf16* w, int stride, int cb, int kb, int lane) {
  const int wrow = cb + (lane & 15);
  const int k0   = kb + ((lane & 16) ? 8 : 0);
  const v8bf* p  = (const v8bf*)(w + wrow * stride + k0);
  return cat8(p[0], p[2]);
}

// A operand tile (M=16 x K=32, bf16) from a row-major [16][H] LDS scratch.
// Lane L holds row L%16; elems 0..7 = K = kc*32+off+0..7, elems 8..15 = +16.
static __device__ inline v16bf loadA_lds(const __bf16* s, int kc, int lane) {
  const int row = lane & 15;
  const int off = (lane & 16) ? 8 : 0;
  const __bf16* p = s + row * H + kc * 32 + off;
  return cat8(*(const v8bf*)(p), *(const v8bf*)(p + 16));
}

// A operand tile of x (f32 global -> bf16), same A layout, direct from HBM.
static __device__ inline v16bf loadA_x(const float* xrow, int kc, int lane) {
  const int off = (lane & 16) ? 8 : 0;
  const float* p = xrow + kc * 32 + off;
  v4f a0 = *(const v4f*)(p);
  v4f a1 = *(const v4f*)(p + 4);
  v4f b0 = *(const v4f*)(p + 16);
  v4f b1 = *(const v4f*)(p + 20);
  v16bf r;
#pragma unroll
  for (int i = 0; i < 4; ++i) {
    r[i]      = (__bf16)a0[i];
    r[4 + i]  = (__bf16)a1[i];
    r[8 + i]  = (__bf16)b0[i];
    r[12 + i] = (__bf16)b1[i];
  }
  return r;
}

static __device__ inline v8f wmma_bf16(v16bf a, v16bf b, v8f c) {
  return __builtin_amdgcn_wmma_f32_16x16x32_bf16(
      /*neg_a=*/false, a, /*neg_b=*/false, b,
      /*c_mod=*/(short)0, c, /*reuse_a=*/false, /*reuse_b=*/false);
}

static __device__ inline float sigf(float x) { return 1.0f / (1.0f + __expf(-x)); }
static __device__ inline float tanhf_(float x) {
  float e = __expf(2.0f * x);
  return 1.0f - 2.0f / (e + 1.0f);
}

__global__ __launch_bounds__(WAVES * 32) void lstm_critic_fused(
    const float* __restrict__ x,    const float* __restrict__ h0,
    const float* __restrict__ c0,   const float* __restrict__ Win,
    const float* __restrict__ bin,  const float* __restrict__ Wih,
    const float* __restrict__ Whh,  const float* __restrict__ bih,
    const float* __restrict__ bhh,  const float* __restrict__ Wout,
    const float* __restrict__ bout, float* __restrict__ out) {
  extern __shared__ char smem[];
  __bf16* w_in   = (__bf16*)(smem + OFF_WIN);
  __bf16* w_ih   = (__bf16*)(smem + OFF_WIH);
  __bf16* w_hh   = (__bf16*)(smem + OFF_WHH);
  float*  b_in_s = (float*)(smem + OFF_BIN);
  float*  bsum_s = (float*)(smem + OFF_BSUM);
  float*  wout_s = (float*)(smem + OFF_WOUT);

  const int tid  = threadIdx.x;
  const int lane = tid & 31;
  const int wid  = tid >> 5;
  __bf16* h_s = (__bf16*)(smem + OFF_SCR + wid * SCR_PER_WAVE);
  __bf16* z_s = h_s + RPW * H;

  // ---- stage weights to LDS (f32 -> bf16), once per workgroup ----
  for (int i = tid; i < H * D; i += WAVES * 32) w_in[i] = (__bf16)Win[i];
  for (int i = tid; i < G * H; i += WAVES * 32) w_ih[i] = (__bf16)Wih[i];
  for (int i = tid; i < G * H; i += WAVES * 32) w_hh[i] = (__bf16)Whh[i];
  for (int i = tid; i < H; i += WAVES * 32) { b_in_s[i] = bin[i]; wout_s[i] = Wout[i]; }
  for (int i = tid; i < G; i += WAVES * 32) bsum_s[i] = bih[i] + bhh[i];
  __syncthreads();

  const int b0    = blockIdx.x * RPWG + wid * RPW;  // first batch row of this wave
  const int col16 = lane & 15;                      // N index within C/D tiles
  const int mhi   = (lane & 16) ? 8 : 0;            // C/D row offset for hi half-wave

  // ---- h/c state in VGPRs, WMMA C/D layout: state[n][r] = (row r+mhi, col n*16+col16)
  v8f hreg[4], creg[4];
#pragma unroll
  for (int n = 0; n < 4; ++n) {
#pragma unroll
    for (int r = 0; r < 8; ++r) {
      const int row = b0 + r + mhi;
      const int col = n * 16 + col16;
      hreg[n][r] = h0[(size_t)row * H + col];
      creg[n][r] = c0[(size_t)row * H + col];
    }
  }
  // prime h scratch with h0 (bf16, row-major [16][H]) for first step's A operand
#pragma unroll
  for (int n = 0; n < 4; ++n)
#pragma unroll
    for (int r = 0; r < 8; ++r)
      h_s[(r + mhi) * H + n * 16 + col16] = (__bf16)hreg[n][r];

  const float  bo     = bout[0];
  const float* xrow0  = x + (size_t)(b0 + col16) * S * D;  // this lane's A-row base
  float* hT_out = out + (size_t)B * S;
  float* cT_out = hT_out + (size_t)B * H;

#pragma unroll 1
  for (int t = 0; t < S; ++t) {
    // A operand of h_prev (16x64 bf16 -> two K=32 tiles) via LDS relayout
    v16bf hA0 = loadA_lds(h_s, 0, lane);
    v16bf hA1 = loadA_lds(h_s, 1, lane);

    // ---- encoder: z = relu(x_t @ W_in^T + b_in), K = D = 128 ----
    const float* xr = xrow0 + (size_t)t * D;
    __builtin_prefetch(xr + D, 0, 0);  // next timestep (global_prefetch_b8)
    v16bf xA[4];
#pragma unroll
    for (int kc = 0; kc < 4; ++kc) xA[kc] = loadA_x(xr, kc, lane);

#pragma unroll
    for (int n = 0; n < 4; ++n) {
      v8f acc = {};
#pragma unroll
      for (int kc = 0; kc < 4; ++kc)
        acc = wmma_bf16(xA[kc], loadB(w_in, D, n * 16, kc * 32, lane), acc);
      const float bz = b_in_s[n * 16 + col16];
#pragma unroll
      for (int r = 0; r < 8; ++r) {
        float v = acc[r] + bz;
        z_s[(r + mhi) * H + n * 16 + col16] = (__bf16)(v > 0.0f ? v : 0.0f);
      }
    }
    v16bf zA0 = loadA_lds(z_s, 0, lane);
    v16bf zA1 = loadA_lds(z_s, 1, lane);

    // ---- gates = z @ W_ih^T + h_prev @ W_hh^T + (b_ih+b_hh); LSTM update ----
#pragma unroll
    for (int n = 0; n < 4; ++n) {
      v8f g4[4];
#pragma unroll
      for (int gg = 0; gg < 4; ++gg) {  // i, f, g, o
        const int cb = gg * 64 + n * 16;
        v8f a = {};
        a = wmma_bf16(zA0, loadB(w_ih, H, cb, 0,  lane), a);
        a = wmma_bf16(zA1, loadB(w_ih, H, cb, 32, lane), a);
        a = wmma_bf16(hA0, loadB(w_hh, H, cb, 0,  lane), a);
        a = wmma_bf16(hA1, loadB(w_hh, H, cb, 32, lane), a);
        g4[gg] = a;
      }
      const float bi = bsum_s[0 * 64 + n * 16 + col16];
      const float bf = bsum_s[1 * 64 + n * 16 + col16];
      const float bg = bsum_s[2 * 64 + n * 16 + col16];
      const float bq = bsum_s[3 * 64 + n * 16 + col16];
#pragma unroll
      for (int r = 0; r < 8; ++r) {
        const float ig = sigf(g4[0][r] + bi);
        const float fg = sigf(g4[1][r] + bf);
        const float gv = tanhf_(g4[2][r] + bg);
        const float og = sigf(g4[3][r] + bq);
        const float c  = fg * creg[n][r] + ig * gv;
        const float h  = og * tanhf_(c);
        creg[n][r] = c;
        hreg[n][r] = h;
        h_s[(r + mhi) * H + n * 16 + col16] = (__bf16)h;  // for t+1 and value head
      }
    }

    // ---- value head: value[b,t] = h_t . W_out + b_out (lanes 0..15, one row each)
    if (lane < 16) {
      float acc = bo;
#pragma unroll
      for (int k = 0; k < H; k += 8) {
        v8bf hv = *(const v8bf*)(h_s + lane * H + k);
#pragma unroll
        for (int i = 0; i < 8; ++i) acc += (float)hv[i] * wout_s[k + i];
      }
      out[(size_t)(b0 + lane) * S + t] = acc;
    }
  }

  // ---- final hT / cT (f32, straight from C/D-layout registers) ----
#pragma unroll
  for (int n = 0; n < 4; ++n)
#pragma unroll
    for (int r = 0; r < 8; ++r) {
      const int row = b0 + r + mhi;
      const int col = n * 16 + col16;
      hT_out[(size_t)row * H + col] = hreg[n][r];
      cT_out[(size_t)row * H + col] = creg[n][r];
    }
}

extern "C" void kernel_launch(void* const* d_in, const int* in_sizes, int n_in,
                              void* d_out, int out_size, void* d_ws, size_t ws_size,
                              hipStream_t stream) {
  (void)in_sizes; (void)n_in; (void)out_size; (void)d_ws; (void)ws_size;
  const float* x    = (const float*)d_in[0];
  const float* h0   = (const float*)d_in[1];
  const float* c0   = (const float*)d_in[2];
  const float* Win  = (const float*)d_in[3];
  const float* bin  = (const float*)d_in[4];
  const float* Wih  = (const float*)d_in[5];
  const float* Whh  = (const float*)d_in[6];
  const float* bih  = (const float*)d_in[7];
  const float* bhh  = (const float*)d_in[8];
  const float* Wout = (const float*)d_in[9];
  const float* bout = (const float*)d_in[10];
  float* out = (float*)d_out;

  dim3 grid(B / RPWG), block(WAVES * 32);
  lstm_critic_fused<<<grid, block, LDS_BYTES, stream>>>(
      x, h0, c0, Win, bin, Wih, Whh, bih, bhh, Wout, bout, out);
}
